// Relevance_weights_acousticFB1_58377195487661
// MI455X (gfx1250) — matrix operand order repaired and checked by reading, not compiled, not run on hardware
//
#include <hip/hip_runtime.h>

typedef __attribute__((ext_vector_type(16))) _Float16 v16h;
typedef __attribute__((ext_vector_type(8)))  float    v8f;

#define NGF    80
#define TLEN   1000
#define SPLICE 10
#define PATCH  21          // real K (padded to 32 for WMMA; B is zero for K>=21)
#define HID    50          // real N (padded to 64 = 4 tiles of 16)
#define NTILES 4
#define TTILES ((TLEN + 15) / 16)   // 63
#define XPAD   1280        // zero-padded LDS row: multiple of 256, >= 992+15+31+1

// sigmoid(x) = 0.5*tanh(0.5*x) + 0.5  -> one V_TANH_F32 instead of exp+rcp
#if __has_builtin(__builtin_amdgcn_tanhf)
#define HAVE_TANH 1
#else
#define HAVE_TANH 0
#endif

__device__ __forceinline__ float fast_sigmoid(float x) {
#if HAVE_TANH
    return __builtin_amdgcn_tanhf(0.5f * x) * 0.5f + 0.5f;
#else
    return __builtin_amdgcn_rcpf(1.0f + __expf(-x));
#endif
}

__global__ __launch_bounds__(256) void relnet_wmma_kernel(
    const float* __restrict__ x,   // [B,1,NGF,TLEN]
    const float* __restrict__ W1,  // [50,21] row-major
    const float* __restrict__ b1,  // [50]
    const float* __restrict__ W2,  // [1,50]
    const float* __restrict__ b2,  // [1]
    float* __restrict__ out,       // [B,TLEN,NGF]
    int B)
{
    __shared__ _Float16 xp[XPAD];
    __shared__ alignas(32) _Float16 W1p[64 * 32];   // zero-padded [N=64][K=32]
    __shared__ float b1p[64];
    __shared__ float w2p[64];

    const int wg   = blockIdx.x;          // b * NGF + f
    const int bb   = wg / NGF;
    const int ff   = wg % NGF;
    const int tid  = threadIdx.x;
    const int lane = tid & 31;
    const int wave = __builtin_amdgcn_readfirstlane(tid >> 5);  // scalar 0..7
    const int nlo  = lane & 15;
    const int half = lane >> 4;           // 0 or 1

    // ---- Stage zero-padded input row into LDS as f16 (read x exactly once) ----
    const float* xrow = x + ((size_t)bb * NGF + ff) * TLEN;
    for (int i = tid; i < XPAD; i += 256) {
        const int tt = i - SPLICE;
        float v = (tt >= 0 && tt < TLEN) ? xrow[tt] : 0.0f;
        xp[i] = (_Float16)v;
    }
    // ---- Stage zero-padded weights (branch-free fragment reads later) ----
    for (int i = tid; i < 64 * 32; i += 256) {
        const int n = i >> 5, k = i & 31;
        float w = (n < HID && k < PATCH) ? W1[n * PATCH + k] : 0.0f;
        W1p[i] = (_Float16)w;
    }
    if (tid < 64) {
        b1p[tid] = (tid < HID) ? b1[tid] : 0.0f;
        w2p[tid] = (tid < HID) ? W2[tid] : 0.0f;
    }
    __syncthreads();

    // ---- W1 B-fragments (32x16 f16 B in contiguous k order, not the ISA table order): lanes 0-15 hold K=0..15,
    //      lanes 16-31 hold K=16..31; N = lane%16. Row slice is 16 contiguous
    //      halfs, 32B-aligned -> vector LDS load, no branches.
    v16h bfrag[NTILES];
    float b1h[NTILES], w2h[NTILES];
    #pragma unroll
    for (int nt = 0; nt < NTILES; ++nt) {
        const int N = nt * 16 + nlo;
        bfrag[nt] = *reinterpret_cast<const v16h*>(&W1p[N * 32 + 16 * half]);
        b1h[nt] = 0.5f * b1p[N];    // pre-halved for the tanh-sigmoid
        w2h[nt] = 0.5f * w2p[N];
    }
    // Constant term of layer-2 (the +0.5*w2 of every hidden sigmoid): seed the
    // accumulator with this lane's share; the 16-lane reduction makes it exact.
#if HAVE_TANH
    const float accInit = w2h[0] + w2h[1] + w2h[2] + w2h[3];
#else
    const float accInit = 0.0f;
#endif
    const float bias2 = b2[0];

    // A layout (16x32 f16): M = lane%16; half 0 -> K 0-15, half 1 -> K 16-31
    //                       (matches the contiguous B fragment order).
    // K >= 21 slots multiply against zero B columns, so garbage there is fine:
    // load unconditionally (one contiguous run of 16 halfs).
    const int hK = half ? 8 : 0;
    const int M  = nlo;

    const int bit0 = lane & 1;
    const int bit1 = (lane >> 1) & 1;
    const int bit2 = (lane >> 2) & 1;
    const int rOut = 4 * bit0 + 2 * bit1 + bit2;   // time row this lane finalizes

    for (int tt = wave; tt < TTILES; tt += 8) {
        const int t0 = tt * 16;

        // ---- Build Toeplitz A fragment from LDS (branch-free) ----
        v16h a;
        #pragma unroll
        for (int v = 0; v < 8; ++v) {
            #pragma unroll
            for (int p = 0; p < 2; ++p) {
                const int K = 16 * half + 2 * v + p;
                a[2 * v + p] = xp[t0 + M + K];
            }
        }

        // ---- Layer 1: issue all 4 WMMAs back-to-back (XDL overlaps VALU) ----
        v8f c[NTILES];
        #pragma unroll
        for (int nt = 0; nt < NTILES; ++nt) {
            v8f z = {};
            c[nt] = __builtin_amdgcn_wmma_f32_16x16x32_f16(
                        false, a, false, bfrag[nt], (short)0, z, false, false);
        }

        // ---- tanh-sigmoid + fold W2, batched 8-wide for TRANS throughput ----
        float acc[8];
        #pragma unroll
        for (int r = 0; r < 8; ++r) acc[r] = accInit;

        #pragma unroll
        for (int nt = 0; nt < NTILES; ++nt) {
#if HAVE_TANH
            float u[8], th[8];
            #pragma unroll
            for (int r = 0; r < 8; ++r)
                u[r] = __builtin_fmaf(0.5f, c[nt][r], b1h[nt]);
            #pragma unroll
            for (int r = 0; r < 8; ++r)
                th[r] = __builtin_amdgcn_tanhf(u[r]);
            #pragma unroll
            for (int r = 0; r < 8; ++r)
                acc[r] = __builtin_fmaf(w2h[nt], th[r], acc[r]);
#else
            #pragma unroll
            for (int r = 0; r < 8; ++r) {
                float h = fast_sigmoid(c[nt][r] + 2.0f * b1h[nt]);
                acc[r] = __builtin_fmaf(2.0f * w2h[nt], h, acc[r]);
            }
#endif
        }

        // ---- Layer 2: transposing butterfly reduction over the 16 lanes of
        //      each half (8 shuffles); sums land scattered so each lane
        //      finishes exactly one time step.
        float v4[4], v2[2], v1;
        {
            float snd[4], g[4];
            #pragma unroll
            for (int j = 0; j < 4; ++j)
                snd[j] = bit0 ? acc[j] : acc[j + 4];
            #pragma unroll
            for (int j = 0; j < 4; ++j)
                g[j] = __shfl_xor(snd[j], 1, 32);
            #pragma unroll
            for (int j = 0; j < 4; ++j)
                v4[j] = (bit0 ? acc[j + 4] : acc[j]) + g[j];
        }
        {
            float snd[2], g[2];
            #pragma unroll
            for (int j = 0; j < 2; ++j)
                snd[j] = bit1 ? v4[j] : v4[j + 2];
            #pragma unroll
            for (int j = 0; j < 2; ++j)
                g[j] = __shfl_xor(snd[j], 2, 32);
            #pragma unroll
            for (int j = 0; j < 2; ++j)
                v2[j] = (bit1 ? v4[j + 2] : v4[j]) + g[j];
        }
        {
            float snd = bit2 ? v2[0] : v2[1];
            float kp  = bit2 ? v2[1] : v2[0];
            v1 = kp + __shfl_xor(snd, 4, 32);
        }
        v1 += __shfl_xor(v1, 8, 32);   // xor-8 stays inside the 16-lane half

        const int time = t0 + rOut + half * 8;
        if ((lane & 8) == 0 && time < TLEN) {
            out[((size_t)bb * TLEN + time) * NGF + ff] = fast_sigmoid(v1 + bias2);
        }
    }
}

extern "C" void kernel_launch(void* const* d_in, const int* in_sizes, int n_in,
                              void* d_out, int out_size, void* d_ws, size_t ws_size,
                              hipStream_t stream) {
    (void)n_in; (void)out_size; (void)d_ws; (void)ws_size;
    const float* x  = (const float*)d_in[0];
    const float* W1 = (const float*)d_in[1];
    const float* b1 = (const float*)d_in[2];
    const float* W2 = (const float*)d_in[3];
    const float* b2 = (const float*)d_in[4];
    float* out = (float*)d_out;

    const int B = in_sizes[0] / (NGF * TLEN);   // 32
    dim3 grid(B * NGF), block(256);
    hipLaunchKernelGGL(relnet_wmma_kernel, grid, block, 0, stream,
                       x, W1, b1, W2, b2, out, B);
}
